// BoxMultiHeadedAttention_41575283425560
// MI455X (gfx1250) — compile-verified, matrix operand order checked by reading, not run on hardware
//
#include <hip/hip_runtime.h>
#include <hip/hip_bf16.h>
#include <cstdint>

// ---------------------------------------------------------------------------
// BoxMultiHeadedAttention for MI455X (gfx1250, wave32, WMMA bf16 + async/TDM)
// B=4, S=1024, D=1024, H=16, DK=64, concat head dim = 128
// ---------------------------------------------------------------------------

typedef __attribute__((ext_vector_type(16))) __bf16 bf16x16;
typedef __attribute__((ext_vector_type(8)))  float  f32x8;
typedef __attribute__((ext_vector_type(4)))  unsigned int u32x4;
typedef __attribute__((ext_vector_type(8)))  int i32x8;
typedef __attribute__((ext_vector_type(4)))  int i32x4;

#define WMMA_BF16(a, b, c) \
  __builtin_amdgcn_wmma_f32_16x16x32_bf16(false, (a), false, (b), (short)0, (c), false, false)

static constexpr int B_  = 4;
static constexpr int S_  = 1024;
static constexpr int D_  = 1024;
static constexpr int H_  = 16;
static constexpr int DK_ = 64;

// LDS offset of a __shared__ object: low 32 bits of the generic address are
// the wave-relative LDS byte address (flat-aperture rule, ISA 10.2).
__device__ __forceinline__ unsigned lds_off_of(const void* p) {
  return (unsigned)(uintptr_t)p;
}

// ---------------------------------------------------------------------------
// TDM: issue a 2-D tensor_load_to_lds (data_size = 2 bytes / bf16).
//   tile_d0: contiguous elements per row, tile_d1: rows
//   stride0: row stride in elements. LDS receives tile_d1 x tile_d0 packed.
// D# packing per CDNA5 ISA 8.3-8.6 (groups 2/3 zero => 2-D tensor).
// This toolchain exposes the 6-arg builtin (clang-23 / therock-10.0 form):
//   (uint32x4 g0, int32x8 g1, int32x4, int32x4, int32x8, i32 cpol)
// ---------------------------------------------------------------------------
__device__ __forceinline__ void tdm_load_2d_bf16(unsigned lds_addr, const void* gptr,
                                                 unsigned tile_d0, unsigned tile_d1,
                                                 unsigned long long stride0) {
  unsigned long long ga = (unsigned long long)(uintptr_t)gptr;
  u32x4 g0;
  g0[0] = 1u;                                            // count=1 (valid user D#)
  g0[1] = lds_addr;                                      // bits 63:32
  g0[2] = (unsigned)(ga & 0xffffffffu);                  // global_addr[31:0]
  g0[3] = (unsigned)((ga >> 32) & 0x01ffffffu) | 0x80000000u;  // addr[56:32] | type=2
  i32x8 g1;
  g1[0] = (int)(1u << 16);                               // data_size=1 (2 bytes)
  g1[1] = (int)((tile_d0 & 0xffffu) << 16);              // tensor_dim0[15:0]
  g1[2] = (int)(((tile_d0 >> 16) & 0xffffu) | ((tile_d1 & 0xffffu) << 16));  // dim0 hi | dim1 lo
  g1[3] = (int)(((tile_d1 >> 16) & 0xffffu) | ((tile_d0 & 0xffffu) << 16));  // dim1 hi | tile_dim0
  g1[4] = (int)(tile_d1 & 0xffffu);                      // tile_dim1 | tile_dim2=0
  g1[5] = (int)(stride0 & 0xffffffffu);                  // tensor_dim0_stride[31:0]
  g1[6] = (int)((stride0 >> 32) & 0xffffu);              // stride hi | dim1_stride lo = 0
  g1[7] = 0;
  i32x4 gz4;
  gz4[0] = 0; gz4[1] = 0; gz4[2] = 0; gz4[3] = 0;
  i32x8 gz8;
  gz8[0] = 0; gz8[1] = 0; gz8[2] = 0; gz8[3] = 0;
  gz8[4] = 0; gz8[5] = 0; gz8[6] = 0; gz8[7] = 0;
  __builtin_amdgcn_tensor_load_to_lds(g0, g1, gz4, gz4, gz8, 0);
}

// ---------------------------------------------------------------------------
// fp32 -> bf16 conversion (optionally re-strided for building [qa|qg] concat)
// ---------------------------------------------------------------------------
__global__ __launch_bounds__(256)
void cvt_f32_bf16(const float* __restrict__ src, __bf16* __restrict__ dst,
                  long long n, int srcRow, long long dstStride, long long dstOff) {
  long long i = (long long)blockIdx.x * blockDim.x + threadIdx.x;
  if (i >= n) return;
  float v = src[i];
  if (srcRow > 0) {
    long long r = i / srcRow;
    long long c = i - r * srcRow;
    dst[r * dstStride + dstOff + c] = (__bf16)v;
  } else {
    dst[i] = (__bf16)v;
  }
}

// ---------------------------------------------------------------------------
// Generic bf16 GEMM:  y[m,n] = sum_k A[m,k] * W[n,k] + bias[n]
//   A: [M x K] row-major bf16 (lda = K), direct global fragment loads.
//   W: [N x K] row-major bf16; the block's 128x32 W tile is staged into LDS
//      with global_load_async_to_lds_b128, double-buffered on ASYNCcnt.
// 256 threads = 8 waves, 4(M) x 2(N); each wave owns a 32x64 f32 accumulator.
// Epilogue modes:
//   0: bf16 head-interleaved  out[((b*H+h)*S+s)*128 + d_off + d]   (Q/K cat)
//   1: bf16 V-transposed      out[((b*H+h)*64+d)*S + s]
//   3: f32 plain              outf[m*1024 + n]                      (gate logits)
//   4: f32 final              outf[m*1024+n] = sigmoid(gate)*val    (d_out)
// ---------------------------------------------------------------------------
__global__ __launch_bounds__(256)
void gemm_bf16(const __bf16* __restrict__ A, int lda,
               const __bf16* __restrict__ W, int ldw,
               const float* __restrict__ bias, int K, int mode,
               __bf16* __restrict__ outb, int d_off,
               const float* __restrict__ gate, float* __restrict__ outf) {
  __shared__ __align__(64) __bf16 Bt[2][128][32];   // double-buffered W tile

  const int tid   = threadIdx.x;
  const int lane  = tid & 31;
  const int wave  = tid >> 5;
  const int lmod  = lane & 15;
  const int lhalf = lane >> 4;
  const int wm = wave >> 1;       // 0..3
  const int wn = wave & 1;        // 0..1
  const int m0 = blockIdx.x * 128 + wm * 32;
  const int n0blk = blockIdx.y * 128;

  f32x8 acc[2][4];
#pragma unroll
  for (int i = 0; i < 2; ++i)
#pragma unroll
    for (int j = 0; j < 4; ++j)
#pragma unroll
      for (int e = 0; e < 8; ++e) acc[i][j][e] = 0.0f;

  // Issue the async staging of one 128x32 W tile into Bt[buf]:
  // 512 x 16B chunks; each thread owns chunks tid and tid+256
  // (per wave: 2 async b128 instructions, 32 lanes x 16B each).
  auto issueB = [&](int buf, int k0) {
#pragma unroll
    for (int c = 0; c < 2; ++c) {
      const int chunk = tid + c * 256;
      const int row   = chunk >> 2;          // 0..127 (64B per row = 4 chunks)
      const int col8  = (chunk & 3) * 8;     // element offset in row
      const unsigned goff = (unsigned)((((size_t)(n0blk + row) * ldw) + k0 + col8) * 2);
      const unsigned loff = lds_off_of(&Bt[buf][row][col8]);
      asm volatile("global_load_async_to_lds_b128 %0, %1, %2"
                   :: "v"(loff), "v"(goff), "s"(W) : "memory");
    }
  };

  const int nk = K / 32;
  issueB(0, 0);
  for (int kt = 0; kt < nk; ++kt) {
    const int k0 = kt * 32;
    if (kt + 1 < nk) {
      issueB((kt + 1) & 1, k0 + 32);
      asm volatile("s_wait_asynccnt 0x2" ::: "memory");   // drain batch kt
    } else {
      asm volatile("s_wait_asynccnt 0x0" ::: "memory");
    }
    __syncthreads();

    const int kk = lhalf * 16;
    // A fragments: lane holds row m0+i*16+lmod, contiguous 16 bf16 along K
    bf16x16 a0 = *(const bf16x16*)(A + (size_t)(m0 + lmod)      * lda + k0 + kk);
    bf16x16 a1 = *(const bf16x16*)(A + (size_t)(m0 + 16 + lmod) * lda + k0 + kk);
    // B fragments from the staged LDS tile
    const __bf16* bt = &Bt[kt & 1][wn * 64][0];
    bf16x16 b0 = *(const bf16x16*)(bt + ( 0 + lmod) * 32 + kk);
    bf16x16 b1 = *(const bf16x16*)(bt + (16 + lmod) * 32 + kk);
    bf16x16 b2 = *(const bf16x16*)(bt + (32 + lmod) * 32 + kk);
    bf16x16 b3 = *(const bf16x16*)(bt + (48 + lmod) * 32 + kk);

    acc[0][0] = WMMA_BF16(a0, b0, acc[0][0]);
    acc[0][1] = WMMA_BF16(a0, b1, acc[0][1]);
    acc[0][2] = WMMA_BF16(a0, b2, acc[0][2]);
    acc[0][3] = WMMA_BF16(a0, b3, acc[0][3]);
    acc[1][0] = WMMA_BF16(a1, b0, acc[1][0]);
    acc[1][1] = WMMA_BF16(a1, b1, acc[1][1]);
    acc[1][2] = WMMA_BF16(a1, b2, acc[1][2]);
    acc[1][3] = WMMA_BF16(a1, b3, acc[1][3]);

    __syncthreads();   // protect Bt[kt&1] before its re-issue next iter
  }

  // Epilogue. C layout: element e -> row m0+i*16+lhalf*8+e, lane lmod -> col.
#pragma unroll
  for (int i = 0; i < 2; ++i) {
#pragma unroll
    for (int j = 0; j < 4; ++j) {
      const int n = n0blk + wn * 64 + j * 16 + lmod;
      const float bn = bias[n];
#pragma unroll
      for (int e = 0; e < 8; ++e) {
        const int m = m0 + i * 16 + lhalf * 8 + e;
        const float val = acc[i][j][e] + bn;
        const int bb = m >> 10, ss = m & 1023;   // m = b*S + s
        const int hh = n >> 6,  dd = n & 63;     // n = h*64 + d
        if (mode == 0) {
          outb[((size_t)(bb * H_ + hh) * S_ + ss) * 128 + d_off + dd] = (__bf16)val;
        } else if (mode == 1) {
          outb[((size_t)(bb * H_ + hh) * DK_ + dd) * S_ + ss] = (__bf16)val;
        } else if (mode == 3) {
          outf[(size_t)m * 1024 + n] = val;
        } else {
          const size_t idx = (size_t)m * 1024 + n;
          const float g = 1.0f / (1.0f + __expf(-gate[idx]));
          outf[idx] = g * val;
        }
      }
    }
  }
}

// ---------------------------------------------------------------------------
// Flash attention over concat heads (d=128), values d=64.
//   Qcat/Kcat: [B,H,S,128] bf16 ; Vt: [B,H,64,S] bf16 ; mask: [B,S,S] int
//   output xb: [B,S,D] bf16 (heads re-interleaved)
// Block = 128 threads (4 waves); each wave owns 16 query rows; grid covers
// (S/64, H, B). K (32x128) and V^T (64x32) tiles are DMA'd into LDS by the
// Tensor Data Mover (wave 0 issues, TENSORcnt-tracked), shared by all waves.
// ---------------------------------------------------------------------------
__global__ __launch_bounds__(128)
void attention_flash(const __bf16* __restrict__ Qcat, const __bf16* __restrict__ Kcat,
                     const __bf16* __restrict__ Vt, const int* __restrict__ mask,
                     __bf16* __restrict__ xb) {
  __shared__ __align__(64) __bf16 Kt[32][128];     // K tile  (8 KB)
  __shared__ __align__(64) __bf16 Vtile[64][32];   // V^T tile (4 KB)
  __shared__ __align__(32) __bf16 pT[4][16][32];   // per-wave P tile (16q x 32kv)

  const int lane  = threadIdx.x & 31;
  const int wave  = threadIdx.x >> 5;
  const int lmod  = lane & 15;
  const int lhalf = lane >> 4;
  const int b = blockIdx.z, h = blockIdx.y;
  const int q0 = blockIdx.x * 64 + wave * 16;

  const __bf16* Qh = Qcat + (size_t)(b * H_ + h) * S_ * 128;
  const __bf16* Kh = Kcat + (size_t)(b * H_ + h) * S_ * 128;
  const __bf16* Vh = Vt   + (size_t)(b * H_ + h) * DK_ * S_;
  const int*    mb = mask + (size_t)b * S_ * S_;

  // Q A-fragments for all 4 k-steps (d = 128), kept resident.
  bf16x16 qA[4];
#pragma unroll
  for (int t = 0; t < 4; ++t)
    qA[t] = *(const bf16x16*)(Qh + (size_t)(q0 + lmod) * 128 + t * 32 + lhalf * 16);

  f32x8 accd[4];
  float rmax[8], lsum[8];
#pragma unroll
  for (int j = 0; j < 4; ++j)
#pragma unroll
    for (int e = 0; e < 8; ++e) accd[j][e] = 0.0f;
#pragma unroll
  for (int e = 0; e < 8; ++e) { rmax[e] = -3.0e38f; lsum[e] = 0.0f; }

  const float sc = 0.0883883476483f;  // 1/sqrt(2*DK) = 1/sqrt(128)
  const unsigned kt_lds = lds_off_of(&Kt[0][0]);
  const unsigned vt_lds = lds_off_of(&Vtile[0][0]);

  for (int kv0 = 0; kv0 < S_; kv0 += 32) {
    // ---- TDM: stage K (32x128) and V^T (64x32) tiles into LDS ----
    if (wave == 0) {
      tdm_load_2d_bf16(kt_lds, Kh + (size_t)kv0 * 128, 128u, 32u, 128ull);
      tdm_load_2d_bf16(vt_lds, Vh + kv0,                32u, 64u, (unsigned long long)S_);
      __builtin_amdgcn_s_wait_tensorcnt(0);
    }
    __syncthreads();

    // ---- scores: S0 covers kv0..kv0+15, S1 covers kv0+16..kv0+31 ----
    f32x8 s0, s1;
#pragma unroll
    for (int e = 0; e < 8; ++e) { s0[e] = 0.0f; s1[e] = 0.0f; }
#pragma unroll
    for (int t = 0; t < 4; ++t) {
      bf16x16 k0v = *(const bf16x16*)&Kt[lmod][t * 32 + lhalf * 16];
      bf16x16 k1v = *(const bf16x16*)&Kt[16 + lmod][t * 32 + lhalf * 16];
      s0 = WMMA_BF16(qA[t], k0v, s0);
      s1 = WMMA_BF16(qA[t], k1v, s1);
    }

    // ---- scale + mask ----
    float p0[8], p1[8], tred[8];
#pragma unroll
    for (int e = 0; e < 8; ++e) {
      const int q = q0 + lhalf * 8 + e;
      float v0 = s0[e] * sc;
      float v1 = s1[e] * sc;
      if (mb[(size_t)q * S_ + kv0 + lmod] == 0)      v0 = -1.0e9f;
      if (mb[(size_t)q * S_ + kv0 + 16 + lmod] == 0) v1 = -1.0e9f;
      p0[e] = v0; p1[e] = v1;
      tred[e] = fmaxf(v0, v1);
    }
    // row max over 16 lanes (halves stay separate under xor 1,2,4,8)
#pragma unroll
    for (int off = 1; off < 16; off <<= 1)
#pragma unroll
      for (int e = 0; e < 8; ++e)
        tred[e] = fmaxf(tred[e], __shfl_xor(tred[e], off, 32));

    float corr[8];
#pragma unroll
    for (int e = 0; e < 8; ++e) {
      const float nm = fmaxf(rmax[e], tred[e]);
      corr[e] = __expf(rmax[e] - nm);
      rmax[e] = nm;
      p0[e] = __expf(p0[e] - nm);
      p1[e] = __expf(p1[e] - nm);
      tred[e] = p0[e] + p1[e];
    }
#pragma unroll
    for (int off = 1; off < 16; off <<= 1)
#pragma unroll
      for (int e = 0; e < 8; ++e)
        tred[e] += __shfl_xor(tred[e], off, 32);
#pragma unroll
    for (int e = 0; e < 8; ++e) lsum[e] = lsum[e] * corr[e] + tred[e];

    // rescale running accumulator
#pragma unroll
    for (int j = 0; j < 4; ++j)
#pragma unroll
      for (int e = 0; e < 8; ++e) accd[j][e] *= corr[e];

    // ---- transpose P (C layout -> A layout) via wave-local LDS tile ----
    __builtin_amdgcn_wave_barrier();
#pragma unroll
    for (int e = 0; e < 8; ++e) {
      pT[wave][lhalf * 8 + e][lmod]      = (__bf16)p0[e];
      pT[wave][lhalf * 8 + e][16 + lmod] = (__bf16)p1[e];
    }
    asm volatile("s_wait_dscnt 0" ::: "memory");
    __builtin_amdgcn_wave_barrier();
    bf16x16 pA = *(const bf16x16*)&pT[wave][lmod][lhalf * 16];
    __builtin_amdgcn_wave_barrier();

    // ---- PV: acc[16q x 64d] += P[16q x 32kv] @ V[32kv x 64d] ----
#pragma unroll
    for (int j = 0; j < 4; ++j) {
      bf16x16 vB = *(const bf16x16*)&Vtile[j * 16 + lmod][lhalf * 16];
      accd[j] = WMMA_BF16(pA, vB, accd[j]);
    }
    __syncthreads();   // all waves done with Kt/Vtile before next TDM issue
  }

  // ---- normalize + store x as [B,S,D] bf16 ----
  float inv[8];
#pragma unroll
  for (int e = 0; e < 8; ++e) inv[e] = (lsum[e] > 0.0f) ? 1.0f / lsum[e] : 0.0f;
#pragma unroll
  for (int j = 0; j < 4; ++j)
#pragma unroll
    for (int e = 0; e < 8; ++e) {
      const int q = q0 + lhalf * 8 + e;
      xb[((size_t)b * S_ + q) * D_ + h * 64 + j * 16 + lmod] = (__bf16)(accd[j][e] * inv[e]);
    }
}

// ---------------------------------------------------------------------------
// Host launch
// ---------------------------------------------------------------------------
extern "C" void kernel_launch(void* const* d_in, const int* in_sizes, int n_in,
                              void* d_out, int out_size, void* d_ws, size_t ws_size,
                              hipStream_t stream) {
  (void)in_sizes; (void)n_in; (void)out_size; (void)ws_size;

  const float* query_g = (const float*)d_in[0];
  const float* key_g   = (const float*)d_in[1];
  const float* query_a = (const float*)d_in[2];
  const float* key_a   = (const float*)d_in[3];
  const float* value_a = (const float*)d_in[4];
  const int*   mask    = (const int*)  d_in[5];
  const float* Wqg = (const float*)d_in[6];  const float* bqg = (const float*)d_in[7];
  const float* Wkg = (const float*)d_in[8];  const float* bkg = (const float*)d_in[9];
  const float* Wqa = (const float*)d_in[10]; const float* bqa = (const float*)d_in[11];
  const float* Wka = (const float*)d_in[12]; const float* bka = (const float*)d_in[13];
  const float* Wva = (const float*)d_in[14]; const float* bva = (const float*)d_in[15];
  const float* Wgate = (const float*)d_in[16]; const float* bgate = (const float*)d_in[17];
  const float* Winfo = (const float*)d_in[18]; const float* binfo = (const float*)d_in[19];
  float* out = (float*)d_out;

  char* ws = (char*)d_ws;
  size_t off = 0;
  auto alloc = [&](size_t bytes) -> char* {
    char* p = ws + off;
    off += (bytes + 255) & ~(size_t)255;
    return p;
  };

  const size_t NX = (size_t)B_ * S_ * D_;       // 4,194,304
  const size_t NW = (size_t)D_ * D_;            // 1,048,576

  __bf16* qg_b = (__bf16*)alloc(NX * 2);
  __bf16* kg_b = (__bf16*)alloc(NX * 2);
  __bf16* qa_b = (__bf16*)alloc(NX * 2);
  __bf16* ka_b = (__bf16*)alloc(NX * 2);
  __bf16* va_b = (__bf16*)alloc(NX * 2);
  __bf16* Wqg_b = (__bf16*)alloc(NW * 2);
  __bf16* Wkg_b = (__bf16*)alloc(NW * 2);
  __bf16* Wqa_b = (__bf16*)alloc(NW * 2);
  __bf16* Wka_b = (__bf16*)alloc(NW * 2);
  __bf16* Wva_b = (__bf16*)alloc(NW * 2);
  __bf16* Wgate_b = (__bf16*)alloc(2 * NW * 2);
  __bf16* Winfo_b = (__bf16*)alloc(NW * 2);
  __bf16* Qcat = (__bf16*)alloc((size_t)B_ * H_ * S_ * 128 * 2);
  __bf16* Kcat = (__bf16*)alloc((size_t)B_ * H_ * S_ * 128 * 2);
  __bf16* Vt   = (__bf16*)alloc((size_t)B_ * H_ * DK_ * S_ * 2);
  __bf16* xb   = (__bf16*)alloc(NX * 2);
  __bf16* cat_b = (__bf16*)alloc((size_t)B_ * S_ * 2 * D_ * 2);
  float*  gateL = (float*)alloc(NX * 4);

  auto cvt = [&](const float* src, __bf16* dst, size_t n, int srcRow,
                 long long dstStride, long long dstOff) {
    unsigned blocks = (unsigned)((n + 255) / 256);
    cvt_f32_bf16<<<dim3(blocks), dim3(256), 0, stream>>>(
        src, dst, (long long)n, srcRow, dstStride, dstOff);
  };

  // bf16 conversions
  cvt(query_g, qg_b, NX, 0, 0, 0);
  cvt(key_g,   kg_b, NX, 0, 0, 0);
  cvt(query_a, qa_b, NX, 0, 0, 0);
  cvt(key_a,   ka_b, NX, 0, 0, 0);
  cvt(value_a, va_b, NX, 0, 0, 0);
  cvt(Wqg, Wqg_b, NW, 0, 0, 0);
  cvt(Wkg, Wkg_b, NW, 0, 0, 0);
  cvt(Wqa, Wqa_b, NW, 0, 0, 0);
  cvt(Wka, Wka_b, NW, 0, 0, 0);
  cvt(Wva, Wva_b, NW, 0, 0, 0);
  cvt(Wgate, Wgate_b, 2 * NW, 0, 0, 0);
  cvt(Winfo, Winfo_b, NW, 0, 0, 0);
  // gate input concat: [query_a | query_g], row stride 2048
  cvt(query_a, cat_b, NX, D_, 2 * D_, 0);
  cvt(query_g, cat_b, NX, D_, 2 * D_, D_);

  const dim3 gGrid(32, 8), gBlk(256);
  // 5 head projections (M=4096, N=1024, K=1024)
  gemm_bf16<<<gGrid, gBlk, 0, stream>>>(qa_b, D_, Wqa_b, D_, bqa, D_, 0, Qcat, 0,  nullptr, nullptr);
  gemm_bf16<<<gGrid, gBlk, 0, stream>>>(qg_b, D_, Wqg_b, D_, bqg, D_, 0, Qcat, 64, nullptr, nullptr);
  gemm_bf16<<<gGrid, gBlk, 0, stream>>>(ka_b, D_, Wka_b, D_, bka, D_, 0, Kcat, 0,  nullptr, nullptr);
  gemm_bf16<<<gGrid, gBlk, 0, stream>>>(kg_b, D_, Wkg_b, D_, bkg, D_, 0, Kcat, 64, nullptr, nullptr);
  gemm_bf16<<<gGrid, gBlk, 0, stream>>>(va_b, D_, Wva_b, D_, bva, D_, 1, Vt,   0,  nullptr, nullptr);

  // attention: grid (S/64, H, B), 4 waves per block
  attention_flash<<<dim3(S_ / 64, H_, B_), dim3(128), 0, stream>>>(Qcat, Kcat, Vt, mask, xb);

  // gate logits (K = 2048), f32
  gemm_bf16<<<gGrid, gBlk, 0, stream>>>(cat_b, 2 * D_, Wgate_b, 2 * D_, bgate, 2 * D_, 3,
                                        nullptr, 0, nullptr, gateL);
  // info GEMM fused with sigmoid(gate) * info -> d_out
  gemm_bf16<<<gGrid, gBlk, 0, stream>>>(xb, D_, Winfo_b, D_, binfo, D_, 4,
                                        nullptr, 0, gateL, out);
}
